// Conv_88304527606176
// MI455X (gfx1250) — matrix operand order repaired and checked by reading, not compiled, not run on hardware
//
#include <hip/hip_runtime.h>
#include <math.h>

#define N_NODES 50000
#define N_EDGES 640000
#define HIDDEN  128

typedef __bf16 bf16;
typedef bf16  bf16x8  __attribute__((ext_vector_type(8)));
typedef bf16  bf16x16 __attribute__((ext_vector_type(16)));
typedef float f32x8   __attribute__((ext_vector_type(8)));
typedef float f32x4   __attribute__((ext_vector_type(4)));

__device__ __forceinline__ f32x8 wmma_bf16(bf16x16 a, bf16x16 b, f32x8 c) {
    // D = A(16x32 bf16) x B(32x16 bf16) + C(16x16 f32)
    return __builtin_amdgcn_wmma_f32_16x16x32_bf16(
        /*neg_a=*/false, a, /*neg_b=*/false, b,
        /*c_mod=*/(short)0, c, /*reuse_a=*/false, /*reuse_b=*/false);
}

// ---------------------------------------------------------------------------
// Prep: weights -> bf16, transposed to column-major Wt[n*H + k] so each lane's
// B fragment (16 consecutive K for one column n) is a contiguous 32B load.
// ---------------------------------------------------------------------------
__global__ void prep_weights(const float* __restrict__ w_pre,
                             const float* __restrict__ w1,
                             const float* __restrict__ w2,
                             bf16* __restrict__ wpre_t,
                             bf16* __restrict__ w1_t,
                             bf16* __restrict__ w2_t) {
    int idx = blockIdx.x * blockDim.x + threadIdx.x;   // 0 .. 3*16384-1
    int which = idx >> 14;
    int e = idx & 16383;
    int k = e >> 7, n = e & 127;
    const float* w = (which == 0) ? w_pre : (which == 1) ? w1 : w2;
    bf16* o       = (which == 0) ? wpre_t : (which == 1) ? w1_t : w2_t;
    o[n * HIDDEN + k] = (bf16)w[k * HIDDEN + n];
}

// Fold BatchNorm (eval) into per-channel scale/shift.
__global__ void prep_bn(const float* __restrict__ g1, const float* __restrict__ be1,
                        const float* __restrict__ m1, const float* __restrict__ v1,
                        const float* __restrict__ g2, const float* __restrict__ be2,
                        const float* __restrict__ m2, const float* __restrict__ v2,
                        float* __restrict__ sc1, float* __restrict__ sh1,
                        float* __restrict__ sc2, float* __restrict__ sh2) {
    int t = threadIdx.x;
    int ch = t & 127;
    if (t < 128) {
        float s = g1[ch] * rsqrtf(v1[ch] + 1e-5f);
        sc1[ch] = s; sh1[ch] = be1[ch] - m1[ch] * s;
    } else {
        float s = g2[ch] * rsqrtf(v2[ch] + 1e-5f);
        sc2[ch] = s; sh2[ch] = be2[ch] - m2[ch] * s;
    }
}

// Zero the scatter accumulator (float4-wide).
__global__ void zero_f32(float* __restrict__ p) {
    int i = blockIdx.x * blockDim.x + threadIdx.x;     // i < NH/4
    f32x4 z = {0.f, 0.f, 0.f, 0.f};
    *(f32x4*)(p + (size_t)i * 4) = z;
}

// ---------------------------------------------------------------------------
// Edge kernel: y[dst] += gelu(x_feat[src] @ W_pre + b_pre) * bases
// 128 edges / block, 8 waves, each wave owns a 16-row stripe, full N=128.
// ---------------------------------------------------------------------------
__global__ void __launch_bounds__(256)
edge_kernel(const float* __restrict__ x_feat,
            const float* __restrict__ bases,
            const int*   __restrict__ src,
            const int*   __restrict__ dst,
            const bf16*  __restrict__ wpre_t,
            const float* __restrict__ b_pre,
            float* __restrict__ y) {
    __shared__ __align__(16) bf16 A[128 * 136];   // padded stride: spreads LDS banks
    __shared__ int s_src[128];
    __shared__ int s_dst[128];

    const int base = blockIdx.x * 128;
    const int tid  = threadIdx.x;

    if (tid < 128) s_src[tid] = src[base + tid];
    else           s_dst[tid - 128] = dst[base + tid - 128];
    __syncthreads();

    // Gather 128 node rows -> LDS as bf16. 2 threads per row, 64 cols each.
    {
        const int row = tid >> 1, half = tid & 1;
        const float* xr = x_feat + (size_t)s_src[row] * HIDDEN + half * 64;
        bf16* ar = A + row * 136 + half * 64;
        #pragma unroll
        for (int i = 0; i < 16; ++i) {
            f32x4 v = *(const f32x4*)(xr + i * 4);
            ar[i * 4 + 0] = (bf16)v.x;
            ar[i * 4 + 1] = (bf16)v.y;
            ar[i * 4 + 2] = (bf16)v.z;
            ar[i * 4 + 3] = (bf16)v.w;
        }
    }
    __syncthreads();

    const int wave = tid >> 5, lane = tid & 31;
    const int l16 = lane & 15, halfl = lane >> 4;
    const int rowbase = wave * 16;

    f32x8 acc[8] = {};
    #pragma unroll
    for (int s = 0; s < 4; ++s) {                     // K steps of 32
        const bf16* ap = A + (rowbase + l16) * 136 + s * 32 + halfl * 16;
        bf16x8 lo = *(const bf16x8*)ap;               // K = s*32 + halfl*16 + 0..7
        bf16x8 hi = *(const bf16x8*)(ap + 8);         // K = s*32 + halfl*16 + 8..15
        bf16x16 a;
        #pragma unroll
        for (int i = 0; i < 8; ++i) { a[i] = lo[i]; a[i + 8] = hi[i]; }
        #pragma unroll
        for (int t = 0; t < 8; ++t) {                 // 8 N-tiles, reuse A fragment
            const bf16* bp = wpre_t + (size_t)(t * 16 + l16) * HIDDEN + s * 32 + halfl * 16;
            bf16x16 b = *(const bf16x16*)bp;
            acc[t] = wmma_bf16(a, b, acc[t]);
        }
    }

    // Epilogue: bias -> exact GELU -> * bases -> scatter-add to y[dst]
    #pragma unroll
    for (int t = 0; t < 8; ++t) {
        const int n = t * 16 + l16;
        const float bias = b_pre[n];
        #pragma unroll
        for (int r = 0; r < 8; ++r) {
            const int m = r + halfl * 8;
            const int row = rowbase + m;
            const int edge = base + row;
            float v = acc[t][r] + bias;
            v = 0.5f * v * (1.0f + erff(v * 0.70710678118654752f));
            v *= bases[(size_t)edge * HIDDEN + n];
            atomicAdd(&y[(size_t)s_dst[row] * HIDDEN + n], v);
        }
    }
}

// x = x_feat + y ; also emit bf16 copy for the next GEMM.
__global__ void add_kernel(const float* __restrict__ x_feat,
                           const float* __restrict__ y,
                           float* __restrict__ x,
                           bf16* __restrict__ xbf) {
    int i = blockIdx.x * blockDim.x + threadIdx.x;    // i < NH/4
    size_t o = (size_t)i * 4;
    f32x4 a = *(const f32x4*)(x_feat + o);
    f32x4 b = *(const f32x4*)(y + o);
    f32x4 s = {a.x + b.x, a.y + b.y, a.z + b.z, a.w + b.w};
    *(f32x4*)(x + o) = s;
    xbf[o + 0] = (bf16)s.x; xbf[o + 1] = (bf16)s.y;
    xbf[o + 2] = (bf16)s.z; xbf[o + 3] = (bf16)s.w;
}

// ---------------------------------------------------------------------------
// Fused FFN GEMM: relu((A @ W + bias) * scale + shift)
// STAGE 1: emit bf16 h. STAGE 2: emit f32 out = xres + h.
// A fragments read straight from global (each reused by 8 N-tiles in-register).
// ---------------------------------------------------------------------------
template <int STAGE>
__global__ void __launch_bounds__(256)
ffn_kernel(const bf16*  __restrict__ Abf,
           const bf16*  __restrict__ Wt,
           const float* __restrict__ bias,
           const float* __restrict__ scale,
           const float* __restrict__ shift,
           const float* __restrict__ xres,
           bf16*  __restrict__ out_bf,
           float* __restrict__ out_f) {
    const int tid = threadIdx.x;
    const int wave = tid >> 5, lane = tid & 31;
    const int l16 = lane & 15, halfl = lane >> 4;
    const int rowbase = blockIdx.x * 128 + wave * 16;

    int arow = rowbase + l16;
    if (arow >= N_NODES) arow = N_NODES - 1;          // clamp loads; stores guarded

    f32x8 acc[8] = {};
    #pragma unroll
    for (int s = 0; s < 4; ++s) {
        const bf16* ap = Abf + (size_t)arow * HIDDEN + s * 32 + halfl * 16;
        bf16x8 lo = *(const bf16x8*)ap;
        bf16x8 hi = *(const bf16x8*)(ap + 8);
        bf16x16 a;
        #pragma unroll
        for (int i = 0; i < 8; ++i) { a[i] = lo[i]; a[i + 8] = hi[i]; }
        #pragma unroll
        for (int t = 0; t < 8; ++t) {
            const bf16* bp = Wt + (size_t)(t * 16 + l16) * HIDDEN + s * 32 + halfl * 16;
            bf16x16 b = *(const bf16x16*)bp;
            acc[t] = wmma_bf16(a, b, acc[t]);
        }
    }

    #pragma unroll
    for (int t = 0; t < 8; ++t) {
        const int n = t * 16 + l16;
        const float bi = bias[n], sc = scale[n], sh = shift[n];
        #pragma unroll
        for (int r = 0; r < 8; ++r) {
            const int row = rowbase + r + halfl * 8;
            if (row < N_NODES) {
                float v = (acc[t][r] + bi) * sc + sh;
                v = fmaxf(v, 0.0f);
                if (STAGE == 1) {
                    out_bf[(size_t)row * HIDDEN + n] = (bf16)v;
                } else {
                    out_f[(size_t)row * HIDDEN + n] = xres[(size_t)row * HIDDEN + n] + v;
                }
            }
        }
    }
}

// ---------------------------------------------------------------------------
extern "C" void kernel_launch(void* const* d_in, const int* in_sizes, int n_in,
                              void* d_out, int out_size, void* d_ws, size_t ws_size,
                              hipStream_t stream) {
    (void)in_sizes; (void)n_in; (void)out_size; (void)ws_size;

    const float* x_feat = (const float*)d_in[0];
    const float* bases  = (const float*)d_in[1];
    const int*   src    = (const int*)d_in[2];
    const int*   dst    = (const int*)d_in[3];
    const float* w_pre  = (const float*)d_in[4];
    const float* b_pre  = (const float*)d_in[5];
    const float* w1     = (const float*)d_in[6];
    const float* b1     = (const float*)d_in[7];
    const float* g1     = (const float*)d_in[8];
    const float* be1    = (const float*)d_in[9];
    const float* m1     = (const float*)d_in[10];
    const float* v1     = (const float*)d_in[11];
    const float* w2     = (const float*)d_in[12];
    const float* b2     = (const float*)d_in[13];
    const float* g2     = (const float*)d_in[14];
    const float* be2    = (const float*)d_in[15];
    const float* m2     = (const float*)d_in[16];
    const float* v2     = (const float*)d_in[17];
    float* out = (float*)d_out;

    const size_t NH = (size_t)N_NODES * HIDDEN;       // 6,400,000
    char* ws = (char*)d_ws;
    float* ws_y   = (float*)ws; ws += NH * 4;
    float* ws_x   = (float*)ws; ws += NH * 4;
    bf16*  ws_xbf = (bf16*)ws;  ws += NH * 2;
    bf16*  ws_hbf = (bf16*)ws;  ws += NH * 2;
    bf16*  wpre_t = (bf16*)ws;  ws += 16384 * 2;
    bf16*  w1_t   = (bf16*)ws;  ws += 16384 * 2;
    bf16*  w2_t   = (bf16*)ws;  ws += 16384 * 2;
    float* sc1 = (float*)ws; ws += 128 * 4;
    float* sh1 = (float*)ws; ws += 128 * 4;
    float* sc2 = (float*)ws; ws += 128 * 4;
    float* sh2 = (float*)ws; ws += 128 * 4;

    prep_weights<<<192, 256, 0, stream>>>(w_pre, w1, w2, wpre_t, w1_t, w2_t);
    prep_bn<<<1, 256, 0, stream>>>(g1, be1, m1, v1, g2, be2, m2, v2, sc1, sh1, sc2, sh2);
    zero_f32<<<(int)(NH / 4 / 256), 256, 0, stream>>>(ws_y);   // 6250 blocks

    edge_kernel<<<N_EDGES / 128, 256, 0, stream>>>(x_feat, bases, src, dst,
                                                   wpre_t, b_pre, ws_y);     // 5000 blocks

    add_kernel<<<(int)(NH / 4 / 256), 256, 0, stream>>>(x_feat, ws_y, ws_x, ws_xbf);

    const int nblk = (N_NODES + 127) / 128;           // 391
    ffn_kernel<1><<<nblk, 256, 0, stream>>>(ws_xbf, w1_t, b1, sc1, sh1,
                                            nullptr, ws_hbf, nullptr);
    ffn_kernel<2><<<nblk, 256, 0, stream>>>(ws_hbf, w2_t, b2, sc2, sh2,
                                            ws_x, nullptr, out);
}
